// GroupAttention_43473658970178
// MI455X (gfx1250) — matrix operand order repaired and checked by reading, not compiled, not run on hardware
//
#include <hip/hip_runtime.h>

typedef __attribute__((ext_vector_type(16))) _Float16     v16h;
typedef __attribute__((ext_vector_type(8)))  float        v8f;
typedef __attribute__((ext_vector_type(4)))  float        v4f;
typedef __attribute__((ext_vector_type(4)))  unsigned int v4u;

#define QKV_STRIDE 264   // halfs: Q|K cols 0..255 (+8 pad)
#define VT_STRIDE  72    // halfs: V^T rows [128][64+8]
#define AT_STRIDE  72    // halfs: attn   [4*64][64+8]
#define Y_STRIDE   136   // halfs: Y      [64][128+8]

#define QKVW_ELEMS (384 * 128)
#define PW_ELEMS   (128 * 128)

__device__ __forceinline__ v8f wmma_f16(v16h a, v16h b, v8f c) {
  return __builtin_amdgcn_wmma_f32_16x16x32_f16(false, a, false, b, (short)0, c,
                                                false, false);
}

__device__ __forceinline__ v8f zero8() {
  v8f z = {0.f, 0.f, 0.f, 0.f, 0.f, 0.f, 0.f, 0.f};
  return z;
}

// A-fragment (16-bit A 16x32 ISA layout): lane half lh holds
// halves[0..7]=K 8*lh.., halves[8..15]=K 16+8*lh..  (rowp = row base of K-step)
__device__ __forceinline__ v16h lds_a(const _Float16* rowp, int lh) {
  union { v16h v; v4u q[2]; } u;
  u.q[0] = *(const v4u*)(rowp + 8 * lh);
  u.q[1] = *(const v4u*)(rowp + 16 + 8 * lh);
  return u.v;
}
// B-fragment (B^T row, f16 source): lane half lh holds K 8*lh..+7, then K 16+8*lh..+7
__device__ __forceinline__ v16h f16_b(const _Float16* rowp, int lh) {
  union { v16h v; v4u q[2]; } u;
  u.q[0] = *(const v4u*)(rowp + 8 * lh);
  u.q[1] = *(const v4u*)(rowp + 16 + 8 * lh);
  return u.v;
}
// Fragments sourced from fp32 global rows, converted f32->f16 in regs.
__device__ __forceinline__ v16h gld_a(const float* rowp, int lh) {
  const v4f* p0 = (const v4f*)(rowp + 8 * lh);
  const v4f* p1 = (const v4f*)(rowp + 16 + 8 * lh);
  v4f f0 = p0[0], f1 = p0[1], f2 = p1[0], f3 = p1[1];
  union { v16h v; _Float16 h[16]; } u;
#pragma unroll
  for (int i = 0; i < 4; ++i) {
    u.h[i]      = (_Float16)f0[i];
    u.h[4 + i]  = (_Float16)f1[i];
    u.h[8 + i]  = (_Float16)f2[i];
    u.h[12 + i] = (_Float16)f3[i];
  }
  return u.v;
}
__device__ __forceinline__ v16h gld_b(const float* rowp, int lh) {
  const v4f* p0 = (const v4f*)(rowp + 8 * lh);
  const v4f* p1 = (const v4f*)(rowp + 16 + 8 * lh);
  v4f f0 = p0[0], f1 = p0[1], f2 = p1[0], f3 = p1[1];
  union { v16h v; _Float16 h[16]; } u;
#pragma unroll
  for (int i = 0; i < 4; ++i) {
    u.h[i]      = (_Float16)f0[i];
    u.h[4 + i]  = (_Float16)f1[i];
    u.h[8 + i]  = (_Float16)f2[i];
    u.h[12 + i] = (_Float16)f3[i];
  }
  return u.v;
}

// B-fragment selector: pre-converted f16 weights (fast path) or f32 + cvt.
template <bool WF16>
__device__ __forceinline__ v16h wload_b(const float* f32row,
                                        const _Float16* f16row, int lh) {
  if constexpr (WF16)
    return f16_b(f16row, lh);
  else
    return gld_b(f32row, lh);
}

// ---- tiny init kernel: convert qkv_w (384x128) and proj_w (128x128) to f16 ----
__global__ void convert_weights_kernel(const float* __restrict__ qkvw,
                                       const float* __restrict__ pw,
                                       _Float16* __restrict__ wbuf) {
  const int i = blockIdx.x * 256 + threadIdx.x;  // grid covers 65536 elements
  if (i < QKVW_ELEMS)
    wbuf[i] = (_Float16)qkvw[i];
  else
    wbuf[i] = (_Float16)pw[i - QKVW_ELEMS];
}

// One workgroup = one (batch, 8x8 window): 64 tokens x 128 channels.
// 256 threads = 8 wave32.
template <bool WF16>
__launch_bounds__(256)
__global__ void group_attn_kernel(const float* __restrict__ X,
                                  const float* __restrict__ QKVW,
                                  const float* __restrict__ QKVB,
                                  const float* __restrict__ PW,
                                  const float* __restrict__ PB,
                                  const _Float16* __restrict__ W16,
                                  float* __restrict__ OUT) {
  __shared__ __align__(16) _Float16 s_qkv[64 * QKV_STRIDE];
  __shared__ __align__(16) _Float16 s_vT[128 * VT_STRIDE];
  __shared__ __align__(16) _Float16 s_at[4 * 64 * AT_STRIDE];
  __shared__ __align__(16) _Float16 s_y[64 * Y_STRIDE];

  const _Float16* QKVW16 = W16;
  const _Float16* PW16 = W16 + QKVW_ELEMS;

  const int lane = threadIdx.x & 31;
  const int wave = threadIdx.x >> 5;
  const int lr = lane & 15;  // row/col selector within tile
  const int lh = lane >> 4;  // K-chunk / row-half selector

  const int b = blockIdx.x / 1920;
  const int g = blockIdx.x % 1920;
  const int gi = g / 80;
  const int gk = g % 80;
  const size_t base = (size_t)b * (size_t)(128 * 192 * 640);
  const int rowc = (gi * 8) * 640 + gk * 8;  // row(t) = rowc + (t>>3)*640 + (t&7)

  // ---------- Phase 1: QKV = X[64,128] @ qkv_w^T + b  (V stored transposed) ----------
  {
    const int m0 = (wave >> 1) * 16;
    const int half = wave & 1;
    const int tok = m0 + lr;
    const float* xrow =
        X + base + (size_t)(rowc + (tok >> 3) * 640 + (tok & 7)) * 128;
    v16h a[4];
#pragma unroll
    for (int ks = 0; ks < 4; ++ks) a[ks] = gld_a(xrow + ks * 32, lh);

    for (int nt = half * 12; nt < half * 12 + 12; ++nt) {
      const int n0 = nt * 16;
      const float* wrow = QKVW + (size_t)(n0 + lr) * 128;
      const _Float16* wrow16 = QKVW16 + (size_t)(n0 + lr) * 128;
      v8f c = zero8();
#pragma unroll
      for (int ks = 0; ks < 4; ++ks)
        c = wmma_f16(a[ks], wload_b<WF16>(wrow + ks * 32, wrow16 + ks * 32, lh),
                     c);
      const float bias = QKVB[n0 + lr];
      if (__builtin_amdgcn_readfirstlane(n0) < 256) {  // Q,K row-major (uniform)
#pragma unroll
        for (int r = 0; r < 8; ++r)
          s_qkv[(m0 + r + 8 * lh) * QKV_STRIDE + n0 + lr] =
              (_Float16)(c[r] + bias);
      } else {  // V transposed: s_vT[col][tok]
#pragma unroll
        for (int r = 0; r < 8; ++r)
          s_vT[(n0 - 256 + lr) * VT_STRIDE + m0 + r + 8 * lh] =
              (_Float16)(c[r] + bias);
      }
    }
  }
  __syncthreads();

  // ---------- Phase 2: logits = (q @ k^T)*scale, row softmax -> f16 attn ----------
  const float scale = 0.17677669529663687f;  // 32^-0.5
  for (int cb = wave; cb < 16; cb += 8) {
    const int h = cb >> 2;
    const int m0 = (cb & 3) * 16;
    const v16h aq = lds_a(&s_qkv[(m0 + lr) * QKV_STRIDE + h * 32], lh);
    v8f ct[4];
#pragma unroll
    for (int nt = 0; nt < 4; ++nt) {
      const v16h bk =
          f16_b(&s_qkv[(nt * 16 + lr) * QKV_STRIDE + 128 + h * 32], lh);
      ct[nt] = wmma_f16(aq, bk, zero8());
    }
    // C layout: VGPR r, this lane -> row m0+r+8*lh, col nt*16+lr.
    // Row reduction = max/sum across the 16-lane half (xor 1,2,4,8 stays in half).
#pragma unroll
    for (int r = 0; r < 8; ++r) {
      float mx = fmaxf(fmaxf(ct[0][r], ct[1][r]), fmaxf(ct[2][r], ct[3][r]));
#pragma unroll
      for (int off = 1; off < 16; off <<= 1)
        mx = fmaxf(mx, __shfl_xor(mx, off, 32));
      float e[4], sum = 0.f;
#pragma unroll
      for (int nt = 0; nt < 4; ++nt) {
        e[nt] = __expf((ct[nt][r] - mx) * scale);
        sum += e[nt];
      }
#pragma unroll
      for (int off = 1; off < 16; off <<= 1) sum += __shfl_xor(sum, off, 32);
      const float inv = 1.0f / sum;
      const int row = m0 + r + 8 * lh;
#pragma unroll
      for (int nt = 0; nt < 4; ++nt)
        s_at[(h * 64 + row) * AT_STRIDE + nt * 16 + lr] =
            (_Float16)(e[nt] * inv);
    }
  }
  __syncthreads();

  // ---------- Phase 3: Y_head = attn[64,64] @ v[64,32] ----------
  for (int cb = wave; cb < 16; cb += 8) {
    const int h = cb >> 2;
    const int m0 = (cb & 3) * 16;
    const _Float16* arow = &s_at[(h * 64 + m0 + lr) * AT_STRIDE];
    const v16h a0 = lds_a(arow, lh);
    const v16h a1 = lds_a(arow + 32, lh);
#pragma unroll
    for (int nt = 0; nt < 2; ++nt) {
      const _Float16* vrow = &s_vT[(h * 32 + nt * 16 + lr) * VT_STRIDE];
      v8f c = zero8();
      c = wmma_f16(a0, f16_b(vrow, lh), c);
      c = wmma_f16(a1, f16_b(vrow + 32, lh), c);
#pragma unroll
      for (int r = 0; r < 8; ++r)
        s_y[(m0 + r + 8 * lh) * Y_STRIDE + h * 32 + nt * 16 + lr] =
            (_Float16)c[r];
    }
  }
  __syncthreads();

  // ---------- Phase 4: OUT = Y[64,128] @ proj_w^T + proj_b ----------
  {
    const int m0 = (wave >> 1) * 16;
    const int half = wave & 1;
    v16h a[4];
#pragma unroll
    for (int ks = 0; ks < 4; ++ks)
      a[ks] = lds_a(&s_y[(m0 + lr) * Y_STRIDE + ks * 32], lh);
#pragma unroll
    for (int nt = half * 4; nt < half * 4 + 4; ++nt) {
      const int n0 = nt * 16;
      const float* wrow = PW + (size_t)(n0 + lr) * 128;
      const _Float16* wrow16 = PW16 + (size_t)(n0 + lr) * 128;
      v8f c = zero8();
#pragma unroll
      for (int ks = 0; ks < 4; ++ks)
        c = wmma_f16(a[ks], wload_b<WF16>(wrow + ks * 32, wrow16 + ks * 32, lh),
                     c);
      const float bias = PB[n0 + lr];
#pragma unroll
      for (int r = 0; r < 8; ++r) {
        const int tok = m0 + r + 8 * lh;
        OUT[base + (size_t)(rowc + (tok >> 3) * 640 + (tok & 7)) * 128 + n0 +
            lr] = c[r] + bias;
      }
    }
  }
}

extern "C" void kernel_launch(void* const* d_in, const int* in_sizes, int n_in,
                              void* d_out, int out_size, void* d_ws,
                              size_t ws_size, hipStream_t stream) {
  const float* x    = (const float*)d_in[0];
  const float* qkvw = (const float*)d_in[1];
  const float* qkvb = (const float*)d_in[2];
  const float* pw   = (const float*)d_in[3];
  const float* pb   = (const float*)d_in[4];
  float* out = (float*)d_out;
  (void)in_sizes; (void)n_in; (void)out_size;

  const int nblocks = 2 * 1920;  // B * (hg*wg) windows
  const size_t ws_need = (size_t)(QKVW_ELEMS + PW_ELEMS) * sizeof(_Float16);

  if (ws_size >= ws_need) {
    _Float16* w16 = (_Float16*)d_ws;
    convert_weights_kernel<<<(QKVW_ELEMS + PW_ELEMS) / 256, 256, 0, stream>>>(
        qkvw, pw, w16);
    group_attn_kernel<true><<<nblocks, 256, 0, stream>>>(x, qkvw, qkvb, pw, pb,
                                                         w16, out);
  } else {
    group_attn_kernel<false><<<nblocks, 256, 0, stream>>>(x, qkvw, qkvb, pw, pb,
                                                          nullptr, out);
  }
}
